// SNN_tracker_model_9646496547409
// MI455X (gfx1250) — matrix operand order repaired and checked by reading, not compiled, not run on hardware
//
#include <hip/hip_runtime.h>

// ---------------------------------------------------------------------------
// SNN tracker for MI455X (gfx1250, wave32, WMMA).
// Dominant cost: [256 x 145920] x [145920 x 256] GEMM. Strategy: convert both
// operands to bf16 once per call (149 MB total -> fits 192 MB L2), then run a
// split-K v_wmma_f32_16x16x32_bf16 GEMM with f32 atomic accumulation.
// Round 2: 32x32 register blocking per wave (4 WMMAs : 8 b128 loads) to halve
// per-flop operand traffic out of L2 and expose WMMA ILP.
// ---------------------------------------------------------------------------

typedef __attribute__((ext_vector_type(16))) __bf16 v16bf;
typedef __attribute__((ext_vector_type(8)))  __bf16 v8bf;
typedef __attribute__((ext_vector_type(8)))  float  v8f;

#define NB 8         // batch
#define NT 32        // timesteps
#define ND 145920    // 2*240*304
#define NH 256       // hidden
#define NM 256       // GEMM rows = NT*NB
#define SPLITK 32    // layer-1 K splits

__device__ __forceinline__ __bf16 f2bf(float f) { return (__bf16)f; }

// ---- fp32 -> bf16, straight elementwise (weights) --------------------------
__global__ void cvt_weight_bf16(const float* __restrict__ src,
                                __bf16* __restrict__ dst, long long n) {
    long long i = ((long long)blockIdx.x * blockDim.x + threadIdx.x) * 4;
    if (i >= n) return;
    float4 v = *(const float4*)(src + i);
    dst[i + 0] = f2bf(v.x);
    dst[i + 1] = f2bf(v.y);
    dst[i + 2] = f2bf(v.z);
    dst[i + 3] = f2bf(v.w);
}

// ---- fp32 spikes [B,T,D] -> bf16 rows ordered m = t*B + b ------------------
__global__ void cvt_spikes_bf16(const float* __restrict__ src,
                                __bf16* __restrict__ dst) {
    long long i = ((long long)blockIdx.x * blockDim.x + threadIdx.x) * 4;
    if (i >= (long long)NB * NT * ND) return;
    int       d   = (int)(i % ND);
    long long row = i / ND;                 // row = b*NT + t
    int b = (int)(row / NT), t = (int)(row % NT);
    long long m = (long long)t * NB + b;
    float4 v = *(const float4*)(src + i);
    __bf16* o = dst + m * ND + d;
    o[0] = f2bf(v.x); o[1] = f2bf(v.y); o[2] = f2bf(v.z); o[3] = f2bf(v.w);
}

// ---- C[m,n] = bias[n]  (accumulator init for split-K atomics) --------------
__global__ void init_bias(float* __restrict__ C, const float* __restrict__ bias,
                          int M, int N) {
    int i = blockIdx.x * blockDim.x + threadIdx.x;
    if (i < M * N) C[i] = bias[i % N];
}

// ---- bf16 WMMA GEMM, split-K, C += A(MxK,row) * B(NxK,row)^T ---------------
// One wave (32 threads) computes a 32x32 output block = 2x2 grid of 16x16
// WMMA tiles; A/B fragments are each reused twice. grid.z = split-K chunks.
// Fragment layouts used here (wave32, 16x16x32 bf16; contiguous K, same order in A and B):
//   A lane: row (lane&15), K elems {h*16 .. h*16+15},        h  = (lane>>4)
//   B lane: col (lane&15), K elems {h*16 .. h*16+15},        h  = (lane>>4)
__global__ void gemm_wmma_bf16(const __bf16* __restrict__ A,
                               const __bf16* __restrict__ Bm,
                               float* __restrict__ C,
                               int M, int N, int K, int ksteps_per_split) {
    const int tn   = blockIdx.x;       // 32-wide N block
    const int tm   = blockIdx.y;       // 32-tall M block
    const int tz   = blockIdx.z;       // split-K index
    const int lane = threadIdx.x;      // 0..31, EXEC all ones (WMMA requirement)

    const int r  = lane & 15;
    const int hi = lane >> 4;          // 0 or 1

    const __bf16* arow0 = A  + (long long)(tm * 32 + r) * K + hi * 16;
    const __bf16* arow1 = arow0 + (long long)16 * K;
    const __bf16* brow0 = Bm + (long long)(tn * 32 + r) * K + hi * 16;
    const __bf16* brow1 = brow0 + (long long)16 * K;

    int k0 = tz * ksteps_per_split * 32;
    int k1 = k0 + ksteps_per_split * 32;
    if (k1 > K) k1 = K;

    v8f acc00 = {}, acc01 = {}, acc10 = {}, acc11 = {};
    for (int k = k0; k < k1; k += 32) {
        v8bf a0lo = *(const v8bf*)(arow0 + k);
        v8bf a0hi = *(const v8bf*)(arow0 + k + 8);
        v8bf a1lo = *(const v8bf*)(arow1 + k);
        v8bf a1hi = *(const v8bf*)(arow1 + k + 8);
        v16bf b0  = *(const v16bf*)(brow0 + k);      // 32B contiguous
        v16bf b1  = *(const v16bf*)(brow1 + k);      // 32B contiguous
        v16bf a0  = __builtin_shufflevector(a0lo, a0hi,
                      0,1,2,3,4,5,6,7,8,9,10,11,12,13,14,15);
        v16bf a1  = __builtin_shufflevector(a1lo, a1hi,
                      0,1,2,3,4,5,6,7,8,9,10,11,12,13,14,15);
        // prefetch next K chunk (global_prefetch_b8)
        __builtin_prefetch((const void*)(arow0 + k + 32), 0, 1);
        __builtin_prefetch((const void*)(arow1 + k + 32), 0, 1);
        __builtin_prefetch((const void*)(brow0 + k + 32), 0, 1);
        __builtin_prefetch((const void*)(brow1 + k + 32), 0, 1);
        // 4 independent WMMAs (no D->A/B hazard, free to co-issue with loads)
        acc00 = __builtin_amdgcn_wmma_f32_16x16x32_bf16(
                    false, a0, false, b0, (short)0, acc00, false, false);
        acc01 = __builtin_amdgcn_wmma_f32_16x16x32_bf16(
                    false, a0, false, b1, (short)0, acc01, false, false);
        acc10 = __builtin_amdgcn_wmma_f32_16x16x32_bf16(
                    false, a1, false, b0, (short)0, acc10, false, false);
        acc11 = __builtin_amdgcn_wmma_f32_16x16x32_bf16(
                    false, a1, false, b1, (short)0, acc11, false, false);
    }

    // C/D layout: VGPR rr, lanes 0-15 -> M=rr, N=lane; lanes 16-31 -> M=rr+8
    const int mb = tm * 32 + hi * 8;
    const int n0 = tn * 32 + r;
#pragma unroll
    for (int rr = 0; rr < 8; ++rr) {
        atomicAdd(&C[(long long)(mb + rr)      * N + n0     ], acc00[rr]);
        atomicAdd(&C[(long long)(mb + rr)      * N + n0 + 16], acc01[rr]);
        atomicAdd(&C[(long long)(mb + 16 + rr) * N + n0     ], acc10[rr]);
        atomicAdd(&C[(long long)(mb + 16 + rr) * N + n0 + 16], acc11[rr]);
    }
}

// ---- snntorch Leaky (reset='subtract'): sequential over T ------------------
__global__ void lif_scan(const float* __restrict__ cur,
                         __bf16* __restrict__ spk, int T, int BH) {
    int j = blockIdx.x * blockDim.x + threadIdx.x;
    if (j >= BH) return;
    float mem = 0.0f;
    for (int t = 0; t < T; ++t) {
        float c     = cur[t * BH + j];
        float reset = (mem > 1.0f) ? 1.0f : 0.0f;
        mem = 0.9f * mem + c - reset;               // THRESH = 1.0
        spk[t * BH + j] = f2bf((mem > 1.0f) ? 1.0f : 0.0f);
    }
}

__device__ __forceinline__ float sigmoidf_(float x) {
    return 1.0f / (1.0f + expf(-x));
}
__device__ __forceinline__ float clamp01_(float x) {
    return fminf(1.0f, fmaxf(0.0f, x));
}

// ---- heads: one wave per (b,t); 7 dots + sigmoid/tanh/bbox -----------------
__global__ void heads_kernel(const __bf16* __restrict__ feat,  // rows m=t*B+b
                             const float* __restrict__ Wd, const float* __restrict__ bd,
                             const float* __restrict__ Wc, const float* __restrict__ bc,
                             const float* __restrict__ Ws, const float* __restrict__ bs,
                             const float* __restrict__ Wv, const float* __restrict__ bv,
                             float* __restrict__ out) {
    const int bt = blockIdx.x;               // b*NT + t (output order [B,T,...])
    const int b  = bt / NT, t = bt % NT;
    const __bf16* f = feat + (long long)(t * NB + b) * NH;
    const int lane = threadIdx.x;

    float sd = 0.f, c0 = 0.f, c1 = 0.f, s0 = 0.f, s1 = 0.f, v0 = 0.f, v1 = 0.f;
    for (int h = lane; h < NH; h += 32) {
        float x = (float)f[h];
        sd += Wd[h] * x;
        c0 += Wc[h] * x;        c1 += Wc[NH + h] * x;
        s0 += Ws[h] * x;        s1 += Ws[NH + h] * x;
        v0 += Wv[h] * x;        v1 += Wv[NH + h] * x;
    }
#pragma unroll
    for (int off = 16; off > 0; off >>= 1) {
        sd += __shfl_xor(sd, off, 32);
        c0 += __shfl_xor(c0, off, 32);  c1 += __shfl_xor(c1, off, 32);
        s0 += __shfl_xor(s0, off, 32);  s1 += __shfl_xor(s1, off, 32);
        v0 += __shfl_xor(v0, off, 32);  v1 += __shfl_xor(v1, off, 32);
    }
    if (lane == 0) {
        float det = sigmoidf_(sd + bd[0]);
        float cx  = sigmoidf_(c0 + bc[0]), cy = sigmoidf_(c1 + bc[1]);
        float w   = sigmoidf_(s0 + bs[0]), hh = sigmoidf_(s1 + bs[1]);
        float vx  = tanhf(v0 + bv[0]),     vy = tanhf(v1 + bv[1]);
        const int n = NB * NT;
        out[bt] = det;                                   // detection [B,T,1]
        float* center = out + n;                         // [B,T,2]
        center[bt * 2 + 0] = cx;  center[bt * 2 + 1] = cy;
        float* size = out + 3 * n;                       // [B,T,2]
        size[bt * 2 + 0] = w;     size[bt * 2 + 1] = hh;
        float* vel = out + 5 * n;                        // [B,T,2]
        vel[bt * 2 + 0] = vx;     vel[bt * 2 + 1] = vy;
        float* bbox = out + 7 * n;                       // [B,T,4]
        bbox[bt * 4 + 0] = clamp01_(cx - 0.5f * w);
        bbox[bt * 4 + 1] = clamp01_(cy - 0.5f * hh);
        bbox[bt * 4 + 2] = clamp01_(cx + 0.5f * w);
        bbox[bt * 4 + 3] = clamp01_(cy + 0.5f * hh);
    }
}

// ---------------------------------------------------------------------------
extern "C" void kernel_launch(void* const* d_in, const int* in_sizes, int n_in,
                              void* d_out, int out_size, void* d_ws, size_t ws_size,
                              hipStream_t stream) {
    const float* spikes = (const float*)d_in[0];
    const float* W1 = (const float*)d_in[1];  const float* b1 = (const float*)d_in[2];
    const float* W2 = (const float*)d_in[3];  const float* b2 = (const float*)d_in[4];
    const float* W3 = (const float*)d_in[5];  const float* b3 = (const float*)d_in[6];
    const float* Wd = (const float*)d_in[7];  const float* bd = (const float*)d_in[8];
    const float* Wc = (const float*)d_in[9];  const float* bc = (const float*)d_in[10];
    const float* Ws = (const float*)d_in[11]; const float* bs = (const float*)d_in[12];
    const float* Wv = (const float*)d_in[13]; const float* bv = (const float*)d_in[14];
    float* out = (float*)d_out;

    const long long NX = (long long)NB * NT * ND;   // 37,355,520 (== NH*ND)

    // workspace carve-up (all offsets 256B-aligned)
    char* ws = (char*)d_ws;
    size_t off = 0;
    __bf16* x_bf  = (__bf16*)(ws + off); off += (size_t)NX * 2;        // 74.7 MB
    __bf16* w1_bf = (__bf16*)(ws + off); off += (size_t)NX * 2;        // 74.7 MB
    __bf16* w2_bf = (__bf16*)(ws + off); off += (size_t)NH * NH * 2;
    __bf16* w3_bf = (__bf16*)(ws + off); off += (size_t)NH * NH * 2;
    float*  cur   = (float* )(ws + off); off += (size_t)NM * NH * 4;
    __bf16* spkA  = (__bf16*)(ws + off); off += (size_t)NM * NH * 2;
    __bf16* spkB  = (__bf16*)(ws + off); off += (size_t)NM * NH * 2;
    __bf16* spkC  = (__bf16*)(ws + off); off += (size_t)NM * NH * 2;
    (void)ws_size;

    const int cvt_blocks_big   = (int)((NX / 4 + 255) / 256);          // 36480
    const int cvt_blocks_small = (NH * NH / 4 + 255) / 256;

    // 1) precision conversion: 300MB fp32 -> 149MB bf16 (fits 192MB L2)
    cvt_weight_bf16<<<cvt_blocks_big,   256, 0, stream>>>(W1, w1_bf, NX);
    cvt_weight_bf16<<<cvt_blocks_small, 256, 0, stream>>>(W2, w2_bf, (long long)NH * NH);
    cvt_weight_bf16<<<cvt_blocks_small, 256, 0, stream>>>(W3, w3_bf, (long long)NH * NH);
    cvt_spikes_bf16<<<cvt_blocks_big,   256, 0, stream>>>(spikes, x_bf);

    // 2) layer 1: split-K WMMA GEMM (32x32 blocks, 32 K-splits -> 2048 waves)
    init_bias<<<(NM * NH + 255) / 256, 256, 0, stream>>>(cur, b1, NM, NH);
    {
        const int ktotal = ND / 32;                                    // 4560
        const int ksteps = (ktotal + SPLITK - 1) / SPLITK;             // 143
        gemm_wmma_bf16<<<dim3(NH / 32, NM / 32, SPLITK), 32, 0, stream>>>(
            x_bf, w1_bf, cur, NM, NH, ND, ksteps);
    }
    lif_scan<<<(NB * NH + 255) / 256, 256, 0, stream>>>(cur, spkA, NT, NB * NH);

    // 3) layer 2
    init_bias<<<(NM * NH + 255) / 256, 256, 0, stream>>>(cur, b2, NM, NH);
    gemm_wmma_bf16<<<dim3(NH / 32, NM / 32, 1), 32, 0, stream>>>(
        spkA, w2_bf, cur, NM, NH, NH, NH / 32);
    lif_scan<<<(NB * NH + 255) / 256, 256, 0, stream>>>(cur, spkB, NT, NB * NH);

    // 4) layer 3
    init_bias<<<(NM * NH + 255) / 256, 256, 0, stream>>>(cur, b3, NM, NH);
    gemm_wmma_bf16<<<dim3(NH / 32, NM / 32, 1), 32, 0, stream>>>(
        spkB, w3_bf, cur, NM, NH, NH, NH / 32);
    lif_scan<<<(NB * NH + 255) / 256, 256, 0, stream>>>(cur, spkC, NT, NB * NH);

    // 5) heads: one wave per (b,t)
    heads_kernel<<<NB * NT, 32, 0, stream>>>(spkC, Wd, bd, Wc, bc, Ws, bs,
                                             Wv, bv, out);
    (void)in_sizes; (void)n_in; (void)out_size;
}
